// TwoStreamNetworkTransferLearning_88837103551444
// MI455X (gfx1250) — compile-verified, matrix operand order checked by reading, not run on hardware
//
#include <hip/hip_runtime.h>
#include <hip/hip_bf16.h>
#include <stdint.h>

// ---------------------------------------------------------------------------
// Types
// ---------------------------------------------------------------------------
typedef __attribute__((ext_vector_type(16))) _Float16 v16h;
typedef __attribute__((ext_vector_type(8)))  _Float16 v8h;
typedef __attribute__((ext_vector_type(8)))  float    v8f;
typedef __attribute__((ext_vector_type(4)))  int      v4i;

#define TBM 128
#define TBN 128
#define TBK 32
#define LDT 40   // LDS row stride in halves (32 + 8 pad) -> 80B, 16B aligned

// ---------------------------------------------------------------------------
// CDNA5 async global->LDS copy (16 bytes), with fallback
// builtin signature (probe-verified): (v4i as1*, v4i as3*, Ii offset, Ii cpol)
// ---------------------------------------------------------------------------
typedef __attribute__((address_space(1))) v4i g_v4i;
typedef __attribute__((address_space(3))) v4i l_v4i;

__device__ __forceinline__ void async_cp16(void* lds_dst, const void* gsrc) {
#if __has_builtin(__builtin_amdgcn_global_load_async_to_lds_b128)
  __builtin_amdgcn_global_load_async_to_lds_b128(
      (g_v4i*)(uintptr_t)gsrc,
      (l_v4i*)(uintptr_t)lds_dst,
      0, 0);
#else
  *(v8h*)lds_dst = *(const v8h*)gsrc;
#endif
}

__device__ __forceinline__ void wait_async_all() {
#if __has_builtin(__builtin_amdgcn_s_wait_asynccnt)
  __builtin_amdgcn_s_wait_asynccnt(0);
#else
  asm volatile("s_wait_asynccnt 0x0" ::: "memory");
#endif
}

// ---------------------------------------------------------------------------
// WMMA implicit-GEMM: C[M][N] = A[M][K] * Bt[N][K]^T, fused scale/shift +
// residual + ReLU epilogue. Writes f32 (Cf) and/or f16 (Ch) outputs.
// Block: 256 threads = 8 waves. Block tile 128x128x32. Wave tile 32x64.
// Staging addresses are hoisted out of the k-loop: each thread owns 2 fixed
// (row, chunk) slots per matrix; only the kt element (half) offset advances.
// ---------------------------------------------------------------------------
__global__ __launch_bounds__(256)
void k_gemm_wmma(const _Float16* __restrict__ A, const _Float16* __restrict__ Bt,
                 const float* __restrict__ gamma, const float* __restrict__ beta,
                 const float* __restrict__ resid,
                 float* __restrict__ Cf, _Float16* __restrict__ Ch,
                 int M, int N, int K, int act)
{
  __shared__ __align__(16) _Float16 sA[2][TBM * LDT];
  __shared__ __align__(16) _Float16 sB[2][TBN * LDT];

  const int tid   = threadIdx.x;
  const int lane  = tid & 31;
  const int wave  = tid >> 5;
  const int waveM = wave >> 1;       // 0..3 -> 32-row strips
  const int waveN = wave & 1;        // 0..1 -> 64-col strips
  const int l15   = lane & 15;
  const int lhi   = lane >> 4;
  const long blockM = (long)blockIdx.y * TBM;
  const long blockN = (long)blockIdx.x * TBN;

  // ---- loop-invariant staging slots: rows row0 and row0+64, chunk ch ----
  const int row0 = tid >> 2;
  const int ch   = (tid & 3) * 8;
  const int d0   = row0 * LDT + ch;          // LDS dst offset (halves)
  const int d1   = (row0 + 64) * LDT + ch;
  long ar0 = blockM + row0;       if (ar0 >= M) ar0 = M - 1;
  long ar1 = blockM + row0 + 64;  if (ar1 >= M) ar1 = M - 1;
  long br0 = blockN + row0;       if (br0 >= N) br0 = N - 1;
  long br1 = blockN + row0 + 64;  if (br1 >= N) br1 = N - 1;
  const _Float16* aSrc0 = A  + ar0 * (long)K + ch;
  const _Float16* aSrc1 = A  + ar1 * (long)K + ch;
  const _Float16* bSrc0 = Bt + br0 * (long)K + ch;
  const _Float16* bSrc1 = Bt + br1 * (long)K + ch;

  v8f acc[2][4];
  for (int i = 0; i < 2; ++i)
    for (int j = 0; j < 4; ++j)
      acc[i][j] = (v8f){0.f, 0.f, 0.f, 0.f, 0.f, 0.f, 0.f, 0.f};

  const int nk = K / TBK;

  auto stage = [&](int buf, int kt) {
    async_cp16(&sA[buf][d0], aSrc0 + kt);
    async_cp16(&sA[buf][d1], aSrc1 + kt);
    async_cp16(&sB[buf][d0], bSrc0 + kt);
    async_cp16(&sB[buf][d1], bSrc1 + kt);
  };

  stage(0, 0);

  for (int t = 0; t < nk; ++t) {
    const int buf = t & 1;
    wait_async_all();
    __syncthreads();
    if (t + 1 < nk) {
      stage(buf ^ 1, (t + 1) * TBK);
      if (t + 2 < nk) {
        __builtin_prefetch(aSrc0 + (t + 2) * TBK, 0, 1);
        __builtin_prefetch(bSrc0 + (t + 2) * TBK, 0, 1);
      }
    }

    v16h aF[2], bF[4];
    for (int rm = 0; rm < 2; ++rm) {
      const int r = waveM * 32 + rm * 16 + l15;
      aF[rm] = *(const v16h*)&sA[buf][r * LDT + lhi * 16];
    }
    for (int cn = 0; cn < 4; ++cn) {
      const int c = waveN * 64 + cn * 16 + l15;
      bF[cn] = *(const v16h*)&sB[buf][c * LDT + lhi * 16];
    }
    for (int rm = 0; rm < 2; ++rm)
      for (int cn = 0; cn < 4; ++cn)
        acc[rm][cn] = __builtin_amdgcn_wmma_f32_16x16x32_f16(
            false, aF[rm], false, bF[cn], (short)0, acc[rm][cn], false, false);

    __syncthreads();
  }

  // epilogue: VGPR i, lanes0-15 -> M=i, lanes16-31 -> M=8+i ; N = lane&15
  for (int rm = 0; rm < 2; ++rm) {
    for (int cn = 0; cn < 4; ++cn) {
      const long mBase = blockM + waveM * 32 + rm * 16 + lhi * 8;
      const long nCol  = blockN + waveN * 64 + cn * 16 + l15;
      if (nCol >= N) continue;
      const float g  = gamma ? gamma[nCol] : 1.0f;
      const float bb = beta  ? beta [nCol] : 0.0f;
      for (int i = 0; i < 8; ++i) {
        const long m = mBase + i;
        if (m >= M) break;
        float v = acc[rm][cn][i] * g + bb;
        if (resid) v += resid[m * (long)N + nCol];
        if (act)   v = v > 0.f ? v : 0.f;
        if (Cf) Cf[m * (long)N + nCol] = v;
        if (Ch) Ch[m * (long)N + nCol] = (_Float16)v;
      }
    }
  }
}

// ---------------------------------------------------------------------------
// Preprocess: NCHW f32 frames -> gray -> 3x3 Gaussian (reflect101) -> NHWC f16
// ---------------------------------------------------------------------------
__device__ __forceinline__ int refl(int i, int n) {
  if (i < 0) i = -i;
  if (i >= n) i = 2 * n - 2 - i;
  return i;
}

__global__ void k_preprocess(const float* __restrict__ frames, _Float16* __restrict__ out,
                             int B, int F, int H, int W, int useF)
{
  const long total = (long)B * useF * H * W;
  const float wk[3] = {0.25f, 0.5f, 0.25f};
  for (long idx = (long)blockIdx.x * blockDim.x + threadIdx.x; idx < total;
       idx += (long)gridDim.x * blockDim.x) {
    const int x = (int)(idx % W);
    const int y = (int)((idx / W) % H);
    const long n = idx / ((long)H * W);
    const long b = n / useF, f = n % useF;
    const float* base = frames + ((b * F + f) * 3) * (long)H * W;
    float acc = 0.f;
    for (int dy = -1; dy <= 1; ++dy) {
      const int yy = refl(y + dy, H);
      for (int dx = -1; dx <= 1; ++dx) {
        const int xx = refl(x + dx, W);
        const long o = (long)yy * W + xx;
        const float gray = 0.299f * base[o] + 0.587f * base[(long)H * W + o]
                         + 0.114f * base[2L * H * W + o];
        acc += wk[dy + 1] * wk[dx + 1] * gray;
      }
    }
    _Float16* d = out + idx * 3;
    d[0] = (_Float16)acc; d[1] = (_Float16)acc; d[2] = (_Float16)acc;
  }
}

// ---------------------------------------------------------------------------
// Weight reorders to Bt layout [O][Kpad] (f16, zero-padded K)
// ---------------------------------------------------------------------------
__global__ void k_wconv2d(const float* __restrict__ w, _Float16* __restrict__ o_,
                          int O, int C, int KH, int KW, int Kpad)
{
  const long total = (long)O * Kpad;
  const int K = KH * KW * C;
  for (long idx = (long)blockIdx.x * blockDim.x + threadIdx.x; idx < total;
       idx += (long)gridDim.x * blockDim.x) {
    const int k = (int)(idx % Kpad);
    const long o = idx / Kpad;
    float v = 0.f;
    if (k < K) {
      const int ci = k % C, kw = (k / C) % KW, kh = k / (C * KW);
      v = w[((o * C + ci) * KH + kh) * (long)KW + kw];
    }
    o_[idx] = (_Float16)v;
  }
}

__global__ void k_wconv3d(const float* __restrict__ w, _Float16* __restrict__ o_,
                          int O, int C, int KD, int KH, int KW, int Kpad)
{
  const long total = (long)O * Kpad;
  const int K = KD * KH * KW * C;
  for (long idx = (long)blockIdx.x * blockDim.x + threadIdx.x; idx < total;
       idx += (long)gridDim.x * blockDim.x) {
    const int k = (int)(idx % Kpad);
    const long o = idx / Kpad;
    float v = 0.f;
    if (k < K) {
      const int ci = k % C;
      const int kw = (k / C) % KW;
      const int kh = (k / (C * KW)) % KH;
      const int kd = k / (C * KW * KH);
      v = w[(((o * C + ci) * KD + kd) * KH + kh) * (long)KW + kw];
    }
    o_[idx] = (_Float16)v;
  }
}

__global__ void k_wflat(const float* __restrict__ w, _Float16* __restrict__ o_,
                        int O, int K, int Kpad)
{
  const long total = (long)O * Kpad;
  for (long idx = (long)blockIdx.x * blockDim.x + threadIdx.x; idx < total;
       idx += (long)gridDim.x * blockDim.x) {
    const int k = (int)(idx % Kpad);
    const long o = idx / Kpad;
    o_[idx] = (_Float16)(k < K ? w[o * K + k] : 0.f);
  }
}

// ---------------------------------------------------------------------------
// im2col (2D and 3D) from NHWC/NDHWC f16, to A[Mc][Kpad] starting at row m0
// ---------------------------------------------------------------------------
__global__ void k_im2col2d(const _Float16* __restrict__ x, _Float16* __restrict__ A,
                           long m0, long Mc, int H, int W, int C,
                           int KH, int KW, int stride, int pad, int OH, int OW, int Kpad)
{
  const long total = Mc * Kpad;
  const int K = KH * KW * C;
  for (long idx = (long)blockIdx.x * blockDim.x + threadIdx.x; idx < total;
       idx += (long)gridDim.x * blockDim.x) {
    const int  k = (int)(idx % Kpad);
    const long m = m0 + idx / Kpad;
    _Float16 v = (_Float16)0.f;
    if (k < K) {
      const int ci = k % C, kw = (k / C) % KW, kh = k / (C * KW);
      const int ow = (int)(m % OW), oh = (int)((m / OW) % OH);
      const long n = m / ((long)OW * OH);
      const int ih = oh * stride - pad + kh, iw = ow * stride - pad + kw;
      if (ih >= 0 && ih < H && iw >= 0 && iw < W)
        v = x[((n * H + ih) * (long)W + iw) * C + ci];
    }
    A[idx] = v;
  }
}

__global__ void k_im2col3d(const _Float16* __restrict__ x, _Float16* __restrict__ A,
                           long m0, long Mc, int D, int H, int W, int C,
                           int KD, int KH, int KW,
                           int sd, int sh, int sw, int pd, int ph, int pw,
                           int OD, int OH, int OW, int Kpad)
{
  const long total = Mc * Kpad;
  const int K = KD * KH * KW * C;
  for (long idx = (long)blockIdx.x * blockDim.x + threadIdx.x; idx < total;
       idx += (long)gridDim.x * blockDim.x) {
    const int  k = (int)(idx % Kpad);
    const long m = m0 + idx / Kpad;
    _Float16 v = (_Float16)0.f;
    if (k < K) {
      const int ci = k % C;
      const int kw = (k / C) % KW;
      const int kh = (k / (C * KW)) % KH;
      const int kd = k / (C * KW * KH);
      const int ow = (int)(m % OW);
      const int oh = (int)((m / OW) % OH);
      const int od = (int)((m / ((long)OW * OH)) % OD);
      const long n = m / ((long)OW * OH * OD);
      const int id_ = od * sd - pd + kd;
      const int ih  = oh * sh - ph + kh;
      const int iw  = ow * sw - pw + kw;
      if (id_ >= 0 && id_ < D && ih >= 0 && ih < H && iw >= 0 && iw < W)
        v = x[(((n * D + id_) * H + ih) * (long)W + iw) * C + ci];
    }
    A[idx] = v;
  }
}

// ---------------------------------------------------------------------------
// 3x3 s2 p1 maxpool, NHWC f16
// ---------------------------------------------------------------------------
__global__ void k_maxpool2d(const _Float16* __restrict__ x, _Float16* __restrict__ y,
                            int N, int H, int W, int C, int OH, int OW)
{
  const long total = (long)N * OH * OW * C;
  for (long idx = (long)blockIdx.x * blockDim.x + threadIdx.x; idx < total;
       idx += (long)gridDim.x * blockDim.x) {
    const int c  = (int)(idx % C);
    const int ow = (int)((idx / C) % OW);
    const int oh = (int)((idx / ((long)C * OW)) % OH);
    const long n = idx / ((long)C * OW * OH);
    float m = -3.4e38f;
    for (int kh = 0; kh < 3; ++kh) {
      const int ih = oh * 2 - 1 + kh;
      if (ih < 0 || ih >= H) continue;
      for (int kw = 0; kw < 3; ++kw) {
        const int iw = ow * 2 - 1 + kw;
        if (iw < 0 || iw >= W) continue;
        const float v = (float)x[((n * H + ih) * (long)W + iw) * C + c];
        m = v > m ? v : m;
      }
    }
    y[idx] = (_Float16)m;
  }
}

// ---------------------------------------------------------------------------
// global average pool over spatial S: [N][S][C] f16 -> f32 + f16 [N][C]
// ---------------------------------------------------------------------------
__global__ void k_gavgpool(const _Float16* __restrict__ x, float* __restrict__ of,
                           _Float16* __restrict__ oh, int N, long S, int C)
{
  const long total = (long)N * C;
  for (long idx = (long)blockIdx.x * blockDim.x + threadIdx.x; idx < total;
       idx += (long)gridDim.x * blockDim.x) {
    const int c  = (int)(idx % C);
    const long n = idx / C;
    float s = 0.f;
    for (long p = 0; p < S; ++p) s += (float)x[(n * S + p) * C + c];
    s /= (float)S;
    of[idx] = s;
    oh[idx] = (_Float16)s;
  }
}

// ---------------------------------------------------------------------------
// Attention head: sfAll[B*24][128] -> mean over pairs -> fc256+tanh -> fc24
// -> softmax -> a[B][24].  One block (256 thr) per batch element.
// ---------------------------------------------------------------------------
__global__ void k_attn(const float* __restrict__ sfAll,
                       const float* __restrict__ w1, const float* __restrict__ b1,
                       const float* __restrict__ w2, const float* __restrict__ b2,
                       float* __restrict__ a)
{
  __shared__ float sf[256], h[256], lg[24];
  const int b = blockIdx.x, t = threadIdx.x;
  {
    float s = 0.f;
    for (int i = 0; i < 12; ++i)
      s += sfAll[((long)b * 24 + 2 * i + (t >> 7)) * 128 + (t & 127)];
    sf[t] = s / 12.f;
  }
  __syncthreads();
  {
    float s = b1[t];
    for (int k = 0; k < 256; ++k) s += w1[(long)t * 256 + k] * sf[k];
    h[t] = tanhf(s);
  }
  __syncthreads();
  if (t < 24) {
    float s = b2[t];
    for (int k = 0; k < 256; ++k) s += w2[(long)t * 256 + k] * h[k];
    lg[t] = s;
  }
  __syncthreads();
  if (t == 0) {
    float mx = lg[0];
    for (int i = 1; i < 24; ++i) mx = lg[i] > mx ? lg[i] : mx;
    float z = 0.f;
    for (int i = 0; i < 24; ++i) { lg[i] = expf(lg[i] - mx); z += lg[i]; }
    for (int i = 0; i < 24; ++i) a[(long)b * 24 + i] = lg[i] / z;
  }
}

// ---------------------------------------------------------------------------
// Gate temporal frames: frames[b, 24+d, c, h, w] * (1+a[b][d]) -> NDHWC f16
// ---------------------------------------------------------------------------
__global__ void k_gate(const float* __restrict__ frames, const float* __restrict__ a,
                       _Float16* __restrict__ y, int B, int F, int H, int W)
{
  const long total = (long)B * 24 * H * W * 3;
  for (long idx = (long)blockIdx.x * blockDim.x + threadIdx.x; idx < total;
       idx += (long)gridDim.x * blockDim.x) {
    const int c  = (int)(idx % 3);
    const int x  = (int)((idx / 3) % W);
    const int yy = (int)((idx / (3L * W)) % H);
    const int d  = (int)((idx / (3L * W * H)) % 24);
    const long b = idx / (3L * W * H * 24);
    const float g = 1.f + a[b * 24 + d];
    const float v = frames[(((b * F + 24 + d) * 3 + c) * (long)H + yy) * W + x];
    y[idx] = (_Float16)(v * g);
  }
}

// ---------------------------------------------------------------------------
// Final: out[b] = sigmoid(dot(tf[b], w) + bias)
// ---------------------------------------------------------------------------
__global__ void k_final(const float* __restrict__ tf, const float* __restrict__ w,
                        const float* __restrict__ b, float* __restrict__ out,
                        int B, int K)
{
  const int i = threadIdx.x;
  if (i < B) {
    float s = b[0];
    for (int k = 0; k < K; ++k) s += tf[(long)i * K + k] * w[k];
    out[i] = 1.f / (1.f + expf(-s));
  }
}

// ---------------------------------------------------------------------------
// Host-side helpers
// ---------------------------------------------------------------------------
static inline int round32(int k) { return (k + 31) & ~31; }

static void launch_gemm(hipStream_t s, const _Float16* A, const _Float16* Bt,
                        const float* g, const float* b, const float* res,
                        float* Cf, _Float16* Ch, long M, int N, int K, int act)
{
  dim3 grid((N + TBN - 1) / TBN, (unsigned)((M + TBM - 1) / TBM));
  k_gemm_wmma<<<grid, 256, 0, s>>>(A, Bt, g, b, res, Cf, Ch, (int)M, N, K, act);
}

#define GS_BLOCKS 2048
#define GS_THREADS 256

extern "C" void kernel_launch(void* const* d_in, const int* in_sizes, int n_in,
                              void* d_out, int out_size, void* d_ws, size_t ws_size,
                              hipStream_t stream) {
  (void)in_sizes; (void)out_size; (void)ws_size;

  // ---- param cursor (sequential walk of flattened params) ----
  int cur = 1;
  auto P = [&](void) -> const float* {
    const float* p = (const float*)d_in[cur % n_in];
    ++cur;
    return p;
  };
  const float* frames = (const float*)d_in[0];

  // ---- workspace bump allocator ----
  char* wsb = (char*)d_ws;
  size_t off = 0;
  auto alloc = [&](size_t bytes) -> void* {
    bytes = (bytes + 255) & ~(size_t)255;
    void* p = wsb + off;
    off += bytes;
    return p;
  };

  const int B = 4, F = 48, H = 112, W = 112;

  // scratch buffers
  _Float16* pre    = (_Float16*)alloc((size_t)96 * H * W * 3 * 2);
  _Float16* wbuf   = (_Float16*)alloc((size_t)16  << 20);
  const size_t ABUF_BYTES = (size_t)128 << 20;
  _Float16* abuf   = (_Float16*)alloc(ABUF_BYTES);
  _Float16* act1h  = (_Float16*)alloc((size_t)301056 * 64 * 2);
  _Float16* pool1h = (_Float16*)alloc((size_t)75264 * 64 * 2);
  _Float16* b1h    = (_Float16*)alloc((size_t)75264 * 64 * 2);
  _Float16* b2h    = (_Float16*)alloc((size_t)75264 * 64 * 2);
  float*    dsF    = (float*)   alloc((size_t)75264 * 256 * 4);
  _Float16* b3h    = (_Float16*)alloc((size_t)75264 * 256 * 2);
  float*    gpF    = (float*)   alloc((size_t)96 * 256 * 4);
  _Float16* gpH    = (_Float16*)alloc((size_t)96 * 256 * 2);
  float*    sfF    = (float*)   alloc((size_t)96 * 128 * 4);
  float*    attA   = (float*)   alloc((size_t)B * 24 * 4);
  _Float16* gateH  = (_Float16*)alloc((size_t)B * 24 * H * W * 3 * 2);
  _Float16* t1h    = (_Float16*)alloc((size_t)301056 * 64 * 2);
  float*    t1F    = (float*)   alloc((size_t)301056 * 64 * 4);
  _Float16* t2h    = (_Float16*)alloc((size_t)301056 * 64 * 2);
  _Float16* t3h    = (_Float16*)alloc((size_t)301056 * 64 * 2);
  float*    tpF    = (float*)   alloc((size_t)B * 64 * 4);
  _Float16* tpH    = (_Float16*)alloc((size_t)B * 64 * 2);
  float*    tfF    = (float*)   alloc((size_t)B * 128 * 4);

  // chunked implicit-GEMM conv2d
  auto conv2d = [&](const _Float16* x, int Nimg, int iH, int iW, int Cin,
                    const float* w, const float* g, const float* bb, const float* res,
                    int Cout, int KH, int KW, int st, int pd,
                    float* outF, _Float16* outH, int act, int* pOH, int* pOW) {
    const int OH = (iH + 2 * pd - KH) / st + 1;
    const int OW = (iW + 2 * pd - KW) / st + 1;
    const int K = KH * KW * Cin, Kpad = round32(K);
    k_wconv2d<<<GS_BLOCKS, GS_THREADS, 0, stream>>>(w, wbuf, Cout, Cin, KH, KW, Kpad);
    const long M = (long)Nimg * OH * OW;
    long rows = (long)(ABUF_BYTES / ((size_t)Kpad * 2)) & ~(long)127;
    if (rows < 128) rows = 128;
    for (long m0 = 0; m0 < M; m0 += rows) {
      const long Mc = (M - m0) < rows ? (M - m0) : rows;
      k_im2col2d<<<GS_BLOCKS, GS_THREADS, 0, stream>>>(x, abuf, m0, Mc, iH, iW, Cin,
                                                       KH, KW, st, pd, OH, OW, Kpad);
      launch_gemm(stream, abuf, wbuf, g, bb,
                  res ? res + m0 * Cout : nullptr,
                  outF ? outF + m0 * Cout : nullptr,
                  outH ? outH + m0 * Cout : nullptr, Mc, Cout, Kpad, act);
    }
    *pOH = OH; *pOW = OW;
  };

  auto conv3d = [&](const _Float16* x, int Nimg, int iD, int iH, int iW, int Cin,
                    const float* w, const float* g, const float* bb, const float* res,
                    int Cout, int KD, int KH, int KW,
                    int sd, int sh, int sw, int pd_, int ph, int pw,
                    float* outF, _Float16* outH, int act,
                    int* pOD, int* pOH, int* pOW) {
    const int OD = (iD + 2 * pd_ - KD) / sd + 1;
    const int OH = (iH + 2 * ph - KH) / sh + 1;
    const int OW = (iW + 2 * pw - KW) / sw + 1;
    const int K = KD * KH * KW * Cin, Kpad = round32(K);
    k_wconv3d<<<GS_BLOCKS, GS_THREADS, 0, stream>>>(w, wbuf, Cout, Cin, KD, KH, KW, Kpad);
    const long M = (long)Nimg * OD * OH * OW;
    long rows = (long)(ABUF_BYTES / ((size_t)Kpad * 2)) & ~(long)127;
    if (rows < 128) rows = 128;
    for (long m0 = 0; m0 < M; m0 += rows) {
      const long Mc = (M - m0) < rows ? (M - m0) : rows;
      k_im2col3d<<<GS_BLOCKS, GS_THREADS, 0, stream>>>(x, abuf, m0, Mc, iD, iH, iW, Cin,
                                                       KD, KH, KW, sd, sh, sw,
                                                       pd_, ph, pw, OD, OH, OW, Kpad);
      launch_gemm(stream, abuf, wbuf, g, bb,
                  res ? res + m0 * Cout : nullptr,
                  outF ? outF + m0 * Cout : nullptr,
                  outH ? outH + m0 * Cout : nullptr, Mc, Cout, Kpad, act);
    }
    *pOD = OD; *pOH = OH; *pOW = OW;
  };

  // =========================== spatial stream ===============================
  k_preprocess<<<GS_BLOCKS, GS_THREADS, 0, stream>>>(frames, pre, B, F, H, W, 24);

  int OH, OW, OD;
  // stem: 3->64, k7 s2 p3 + BN + ReLU  (112 -> 56)
  {
    const float *sw = P(), *sg = P(), *sb = P();
    conv2d(pre, 96, H, W, 3, sw, sg, sb, nullptr, 64, 7, 7, 2, 3,
           nullptr, act1h, 1, &OH, &OW);
  }
  // maxpool 3x3 s2 p1 (56 -> 28)
  const int PH = (OH - 1) / 2 + 1, PW = (OW - 1) / 2 + 1;
  k_maxpool2d<<<GS_BLOCKS, GS_THREADS, 0, stream>>>(act1h, pool1h, 96, OH, OW, 64, PH, PW);

  // bottleneck block (64 -> 64 -> 64 -> 256, + 1x1 downsample residual)
  {
    const float *w1 = P(), *g1 = P(), *bb1 = P();
    const float *w2 = P(), *g2 = P(), *bb2 = P();
    const float *w3 = P(), *g3 = P(), *bb3 = P();
    const float *wd = P(), *gd = P(), *bd = P();
    int oh, ow;
    // 1x1: direct GEMM on activation matrix (K = Cin, already mult of 32)
    k_wflat<<<GS_BLOCKS, GS_THREADS, 0, stream>>>(w1, wbuf, 64, 64, 64);
    launch_gemm(stream, pool1h, wbuf, g1, bb1, nullptr, nullptr, b1h,
                (long)96 * PH * PW, 64, 64, 1);
    // 3x3 s1 p1
    conv2d(b1h, 96, PH, PW, 64, w2, g2, bb2, nullptr, 64, 3, 3, 1, 1,
           nullptr, b2h, 1, &oh, &ow);
    // downsample 1x1 64->256 (f32 out, used as residual)
    k_wflat<<<GS_BLOCKS, GS_THREADS, 0, stream>>>(wd, wbuf, 256, 64, 64);
    launch_gemm(stream, pool1h, wbuf, gd, bd, nullptr, dsF, nullptr,
                (long)96 * PH * PW, 256, 64, 0);
    // 1x1 64->256 + residual + ReLU
    k_wflat<<<GS_BLOCKS, GS_THREADS, 0, stream>>>(w3, wbuf, 256, 64, 64);
    launch_gemm(stream, b2h, wbuf, g3, bb3, dsF, nullptr, b3h,
                (long)96 * PH * PW, 256, 64, 1);
  }

  // global avg pool -> (96, 256)
  k_gavgpool<<<GS_BLOCKS, GS_THREADS, 0, stream>>>(b3h, gpF, gpH, 96, (long)PH * PW, 256);

  // spatial FC -> (96, 128)
  {
    const float *fw = P(), *fb = P();
    k_wflat<<<GS_BLOCKS, GS_THREADS, 0, stream>>>(fw, wbuf, 128, 256, 256);
    launch_gemm(stream, gpH, wbuf, nullptr, fb, nullptr, sfF, nullptr, 96, 128, 256, 0);
  }

  // attention -> a[4][24]
  {
    const float *aw1 = P(), *ab1 = P(), *aw2 = P(), *ab2 = P();
    k_attn<<<B, 256, 0, stream>>>(sfF, aw1, ab1, aw2, ab2, attA);
  }

  // gate temporal frames -> NDHWC f16 (4, 24, 112, 112, 3)
  k_gate<<<GS_BLOCKS, GS_THREADS, 0, stream>>>(frames, attA, gateH, B, F, H, W);

  // =========================== temporal stream ==============================
  {
    const float *tw = P(), *tg = P(), *tb = P();
    conv3d(gateH, B, 24, H, W, 3, tw, tg, tb, nullptr, 64, 7, 7, 7,
           1, 2, 2, 3, 3, 3, t1F, t1h, 1, &OD, &OH, &OW);
  }
  // basic3d block: conv1 + ReLU, conv2 + residual(t1) + ReLU
  {
    const float *w1 = P(), *g1 = P(), *bb1 = P();
    const float *w2 = P(), *g2 = P(), *bb2 = P();
    int od, oh, ow;
    conv3d(t1h, B, OD, OH, OW, 64, w1, g1, bb1, nullptr, 64, 3, 3, 3,
           1, 1, 1, 1, 1, 1, nullptr, t2h, 1, &od, &oh, &ow);
    conv3d(t2h, B, OD, OH, OW, 64, w2, g2, bb2, t1F, 64, 3, 3, 3,
           1, 1, 1, 1, 1, 1, nullptr, t3h, 1, &od, &oh, &ow);
  }
  // global avg pool -> (4, 64)
  k_gavgpool<<<GS_BLOCKS, GS_THREADS, 0, stream>>>(t3h, tpF, tpH, B,
                                                   (long)OD * OH * OW, 64);
  // temporal FC -> (4, 128)
  {
    const float *fw = P(), *fb = P();
    k_wflat<<<GS_BLOCKS, GS_THREADS, 0, stream>>>(fw, wbuf, 128, 64, 64);
    launch_gemm(stream, tpH, wbuf, nullptr, fb, nullptr, tfF, nullptr, B, 128, 64, 0);
  }
  // final sigmoid -> d_out (4 floats)
  {
    const float *fw = P(), *fb = P();
    k_final<<<1, 32, 0, stream>>>(tfF, fw, fb, (float*)d_out, B, 128);
  }
}
